// ObjectChecklistModel_75737453297863
// MI455X (gfx1250) — matrix operand order repaired and checked by reading, not compiled, not run on hardware
//
#include <hip/hip_runtime.h>
#include <math.h>

typedef _Float16 h16;
typedef __attribute__((ext_vector_type(16))) _Float16 v16h;
typedef __attribute__((ext_vector_type(8)))  float    v8f;

#define WMMA_F16F32(a,b,c) __builtin_amdgcn_wmma_f32_16x16x32_f16(false,(a),false,(b),(short)0,(c),false,false)

static constexpr int cB  = 32;    // batch
static constexpr int cS  = 256;   // S_IN
static constexpr int cSO = 32;    // S_OUT
static constexpr int cY  = 512;   // Y
static constexpr int cH  = 256;   // H
static constexpr int cE  = 128;   // E
static constexpr int c3H = 768;
static constexpr int cDD = 1504;
static constexpr int cBY = cB * cY;   // 16384

// ---------------- WMMA fragment loaders (wave32, 16x16x32 f16) ----------------
// A 16x32: lanes 0-15 rows 0-15; lane half selects K 0-15 / 16-31, contiguous in K
// (matches frag_b_nt).
__device__ __forceinline__ v16h frag_a(const h16* A, int lda, int m0, int k0, int lane) {
  const h16* p = A + (size_t)(m0 + (lane & 15)) * lda;
  int kb  = k0 + ((lane >> 4) << 4);
  int kb2 = kb + 8;
  v16h a;
#pragma unroll
  for (int i = 0; i < 8; ++i) a[i] = p[kb + i];
#pragma unroll
  for (int i = 0; i < 8; ++i) a[8 + i] = p[kb2 + i];
  return a;
}
// B 32x16 from weight stored [N,K] row-major (computes x@W^T): lane col = lane&15,
// lane half selects K 0-15 / 16-31, contiguous in K (2x b128 per fragment).
__device__ __forceinline__ v16h frag_b_nt(const h16* W, int ldw, int n0, int k0, int lane) {
  const h16* p = W + (size_t)(n0 + (lane & 15)) * ldw + k0 + ((lane >> 4) << 4);
  v16h b;
#pragma unroll
  for (int i = 0; i < 16; ++i) b[i] = p[i];
  return b;
}

__device__ __forceinline__ float wave_max(float v) {
  for (int o = 16; o; o >>= 1) v = fmaxf(v, __shfl_xor(v, o, 32));
  return v;
}
__device__ __forceinline__ float wave_sum(float v) {
  for (int o = 16; o; o >>= 1) v += __shfl_xor(v, o, 32);
  return v;
}

// ---------------- generic NT WMMA GEMM with LDS-staged A tile ----------------
// C[m,n] = sum_k A[m,k]*W[n,k] (+bias[n]) (+=C if ACCUM) (tanh if ACT)
// Requirements (all call sites satisfy): M %16==0, N %128==0, K %32==0, K<=512.
template <int ACCUM, int ACT>
__global__ __launch_bounds__(256) void gemm_nt_k(
    const h16* __restrict__ A, int lda, long long sA,
    const h16* __restrict__ W, int ldw, long long sW,
    float* __restrict__ C, int ldc, long long sC,
    const float* __restrict__ bias, int M, int N, int K) {
  __shared__ h16 As[16 * 512];                 // 16KB max A tile (16 rows x K)
  A += sA * blockIdx.z; W += sW * blockIdx.z; C += sC * blockIdx.z;
  int m0 = blockIdx.y * 16;
  const h16* Arow = A + (size_t)m0 * lda;
  for (int i = threadIdx.x; i < 16 * K; i += 256) {
    int r = i / K, c = i - r * K;
    As[i] = Arow[(size_t)r * lda + c];
  }
  __syncthreads();
  int lane = threadIdx.x & 31, wave = threadIdx.x >> 5;
  int n0 = (blockIdx.x * 8 + wave) * 16;
  v8f acc = {};
#pragma unroll 2
  for (int k0 = 0; k0 < K; k0 += 32) {
    v16h a = frag_a(As, K, 0, k0, lane);
    v16h b = frag_b_nt(W, ldw, n0, k0, lane);
    acc = WMMA_F16F32(a, b, acc);
  }
  int col = n0 + (lane & 15);
  float bv = bias ? bias[col] : 0.0f;
#pragma unroll
  for (int v = 0; v < 8; ++v) {
    int row = m0 + v + ((lane >> 4) << 3);
    float x = acc[v] + bv;
    if (ACCUM) x += C[(size_t)row * ldc + col];
    if (ACT == 1) x = tanhf(x);
    C[(size_t)row * ldc + col] = x;
  }
}

// ---------------- small elementwise / gather kernels ----------------
__global__ void zero2_k(float* a) { if (threadIdx.x < 2) a[threadIdx.x] = 0.0f; }

__global__ void f32tof16_k(const float* __restrict__ s, h16* __restrict__ d, int n) {
  int i = blockIdx.x * 256 + threadIdx.x;
  if (i < n) d[i] = (h16)s[i];
}

// src [R,C] f32 -> dst [C,R] f16
__global__ void transpose16_k(const float* __restrict__ src, h16* __restrict__ dst,
                              int R, int Cc) {
  int i = blockIdx.x * 256 + threadIdx.x;
  if (i >= R * Cc) return;
  int r = i / Cc, c = i - r * Cc;
  dst[(size_t)c * R + r] = (h16)src[i];
}

__global__ void gather16_k(const float* __restrict__ tab, const int* __restrict__ idx,
                           h16* __restrict__ d, int rows, int D) {
  int i = blockIdx.x * 256 + threadIdx.x;
  if (i >= rows * D) return;
  d[i] = (h16)tab[(size_t)idx[i / D] * D + (i % D)];
}

// emb_f32[B,S,E]; xs[s,b,e]=emb[b,s,e]; xsr[s,b,e]=emb[b,S-1-s,e]
__global__ void build_emb_k(const float* __restrict__ iemb, const int* __restrict__ x,
                            float* __restrict__ emb, h16* __restrict__ xs,
                            h16* __restrict__ xsr) {
  int i = blockIdx.x * 256 + threadIdx.x;
  if (i >= cB * cS * cE) return;
  int e = i & (cE - 1), s = (i >> 7) & (cS - 1), b = i >> 15;
  float v = iemb[(size_t)x[b * cS + s] * cE + e];
  emb[i] = v;
  xs [((size_t)s * cB + b) * cE + e] = (h16)v;
  xsr[((size_t)(cS - 1 - s) * cB + b) * cE + e] = (h16)v;
}

// persistent bi-GRU scan: blockIdx.x = direction; gi precomputed [S,B,3H]
__global__ __launch_bounds__(256) void enc_scan_k(
    const float* __restrict__ giF, const float* __restrict__ giB,
    const h16* __restrict__ WhhF, const h16* __restrict__ WhhB,
    const float* __restrict__ bhhF, const float* __restrict__ bhhB,
    float* __restrict__ ofF, float* __restrict__ ofB) {
  const float* gi  = blockIdx.x ? giB  : giF;
  const h16*   Whh = blockIdx.x ? WhhB : WhhF;
  const float* bhh = blockIdx.x ? bhhB : bhhF;
  float*       oh  = blockIdx.x ? ofB  : ofF;
  __shared__ h16 hbuf[2][cB * cH];   // 32KB ping-pong
  for (int i = threadIdx.x; i < cB * cH; i += 256) hbuf[0][i] = (h16)0.0f;
  __syncthreads();
  int lane = threadIdx.x & 31, wave = threadIdx.x >> 5;
  for (int t = 0; t < cS; ++t) {
    const h16* hc = hbuf[t & 1];
    h16* hn = hbuf[(t + 1) & 1];
    const float* git = gi + (size_t)t * cB * c3H;
    float* oht = oh + (size_t)t * cB * cH;
#pragma unroll 1
    for (int q = 0; q < 4; ++q) {               // 8 waves * 4 = 32 (mt,jt) triples
      int p = wave * 4 + q;
      int m0 = (p >> 4) * 16, j0 = (p & 15) * 16;
      v8f ar = {}, az = {}, an_ = {};
#pragma unroll 1
      for (int k0 = 0; k0 < cH; k0 += 32) {
        v16h a = frag_a(hc, cH, m0, k0, lane);
        ar  = WMMA_F16F32(a, frag_b_nt(Whh, cH, j0,          k0, lane), ar);
        az  = WMMA_F16F32(a, frag_b_nt(Whh, cH, cH + j0,     k0, lane), az);
        an_ = WMMA_F16F32(a, frag_b_nt(Whh, cH, 2 * cH + j0, k0, lane), an_);
      }
      int j = j0 + (lane & 15);
      float br = bhh[j], bz = bhh[cH + j], bn = bhh[2 * cH + j];
#pragma unroll
      for (int v = 0; v < 8; ++v) {
        int m = m0 + v + ((lane >> 4) << 3);
        float ir = git[m * c3H + j];
        float iz = git[m * c3H + cH + j];
        float in_ = git[m * c3H + 2 * cH + j];
        float r  = 1.0f / (1.0f + expf(-(ir + ar[v] + br)));
        float z  = 1.0f / (1.0f + expf(-(iz + az[v] + bz)));
        float nn = tanhf(in_ + r * (an_[v] + bn));
        float ho = (float)hc[m * cH + j];
        float hv = (1.0f - z) * nn + z * ho;
        hn[m * cH + j] = (h16)hv;
        oht[m * cH + j] = hv;
      }
    }
    __syncthreads();
  }
}

// enc_out[b,s,0:H]=ofF[s,b]; [H:2H]=ofB[S-1-s,b]; also transposed copy [B,2H,S]
__global__ void pack_enc_k(const float* __restrict__ ofF, const float* __restrict__ ofB,
                           h16* __restrict__ eo16, h16* __restrict__ eot16) {
  int i = blockIdx.x * 256 + threadIdx.x;
  if (i >= cB * cS * 2 * cH) return;
  int h = i & 511, s = (i >> 9) & 255, b = i >> 17;
  float v = (h < cH) ? ofF[((size_t)s * cB + b) * cH + h]
                     : ofB[((size_t)(cS - 1 - s) * cB + b) * cH + (h - cH)];
  eo16[i] = (h16)v;
  eot16[((size_t)b * 2 * cH + h) * cS + s] = (h16)v;
}

__global__ void comb_k(const float* __restrict__ emb, const float* __restrict__ tr,
                       h16* __restrict__ comb) {
  int i = blockIdx.x * 256 + threadIdx.x;
  if (i < cB * cS * cE) comb[i] = (h16)(emb[i] + tr[i]);
}

__global__ void softmax_rows_k(const float* __restrict__ lg, h16* __restrict__ al,
                               int rows, int len) {
  int lane = threadIdx.x & 31, wave = threadIdx.x >> 5;
  int row = blockIdx.x * 8 + wave;
  if (row >= rows) return;
  const float* p = lg + (size_t)row * len;
  float mx = -1e30f;
  for (int i = lane; i < len; i += 32) mx = fmaxf(mx, p[i]);
  mx = wave_max(mx);
  float s = 0.0f;
  for (int i = lane; i < len; i += 32) s += expf(p[i] - mx);
  s = wave_sum(s);
  float inv = 1.0f / s;
  h16* q = al + (size_t)row * len;
  for (int i = lane; i < len; i += 32) q[i] = (h16)(expf(p[i] - mx) * inv);
}

// torch contiguous-view merge: h0[i,j]=tanh(sum_k hiddenflat[i*512+k]*W[j,k]+b[j])
__global__ void merge_k(const float* __restrict__ hfF, const float* __restrict__ hfB,
                        const float* __restrict__ W, const float* __restrict__ bb,
                        float* __restrict__ h0) {
  int i = blockIdx.x, j = threadIdx.x;
  float acc = bb[j];
  for (int k = 0; k < 2 * cH; ++k) {
    int f = i * 2 * cH + k;
    int d = f >> 13, rem = f & 8191, b = rem >> 8, h = rem & 255;
    float hv = (d == 0 ? hfF : hfB)[b * cH + h];
    acc += hv * W[j * 2 * cH + k];
  }
  h0[i * cH + j] = tanhf(acc);
}

__global__ void hinit_k(const float* __restrict__ h0, h16* __restrict__ hd) {
  int i = blockIdx.x * 256 + threadIdx.x;
  if (i >= cBY * cH) return;
  int m = i >> 8, j = i & 255;
  hd[i] = (h16)h0[(m >> 9) * cH + j];
}

__global__ void atb_k(const h16* __restrict__ at, h16* __restrict__ atb) {
  int i = blockIdx.x * 256 + threadIdx.x;
  if (i >= cBY * 32) return;
  int m = i >> 5, c = i & 31;
  atb[i] = at[(m & 511) * 32 + c];
}

__global__ void yp_k(const float* __restrict__ y, h16* __restrict__ yp) {
  int i = blockIdx.x * 256 + threadIdx.x;
  if (i >= cB * cSO * cY) return;
  int yy = i & 511, r = i >> 9, t = r & 31, b = r >> 5;
  yp[i] = (t == 0) ? (h16)0.0f : (h16)y[((size_t)(b * cSO + t - 1)) * cY + yy];
}

__global__ void projcnt_k(const float* __restrict__ ct, const float* __restrict__ W,
                          float* __restrict__ pc) {
  int i = blockIdx.x * 256 + threadIdx.x;
  if (i >= 4 * c3H) return;
  int r = i / c3H, n = i % c3H;
  float a = 0.0f;
  for (int k = 0; k < 32; ++k) a += ct[r * 32 + k] * W[(size_t)n * cDD + 512 + k];
  pc[i] = a;
}

// fused decoder step: gh WMMA + gate nonlinearity in accumulator registers.
// Block = one 16-row h tile (staged in LDS) x 8 jt triples; 2 blocks per m-tile.
__global__ __launch_bounds__(256) void dec_step_k(
    const h16* __restrict__ hc, h16* __restrict__ hn,
    const h16* __restrict__ Whh, const float* __restrict__ bhh,
    const float* __restrict__ giC, const float* __restrict__ pc,
    const float* __restrict__ rp, const float* __restrict__ yd,
    const int* __restrict__ cf, int t) {
  __shared__ h16 As[16 * cH];                  // 8KB staged h tile
  int m0 = (blockIdx.x >> 1) * 16;
  int jbase = (blockIdx.x & 1) * 8;
  {
    const float4* s4 = (const float4*)(hc + (size_t)m0 * cH);
    float4* d4 = (float4*)As;
    d4[threadIdx.x * 2]     = s4[threadIdx.x * 2];
    d4[threadIdx.x * 2 + 1] = s4[threadIdx.x * 2 + 1];
  }
  __syncthreads();
  int lane = threadIdx.x & 31, wave = threadIdx.x >> 5;
  int j0 = (jbase + wave) * 16;
  v8f ar = {}, az = {}, an_ = {};
#pragma unroll 1
  for (int k0 = 0; k0 < cH; k0 += 32) {
    v16h a = frag_a(As, cH, 0, k0, lane);
    ar  = WMMA_F16F32(a, frag_b_nt(Whh, cH, j0,          k0, lane), ar);
    az  = WMMA_F16F32(a, frag_b_nt(Whh, cH, cH + j0,     k0, lane), az);
    an_ = WMMA_F16F32(a, frag_b_nt(Whh, cH, 2 * cH + j0, k0, lane), an_);
  }
  int j = j0 + (lane & 15);
  float br = bhh[j], bz = bhh[cH + j], bn = bhh[2 * cH + j];
#pragma unroll
  for (int v = 0; v < 8; ++v) {
    int lr = v + ((lane >> 4) << 3);
    int m = m0 + lr;
    int b = m >> 9, y = m & 511;
    const float* gc  = giC + (size_t)m * c3H;
    const float* pcr = pc + (size_t)cf[(b * cSO + t) * cY + y] * c3H;
    const float* rpr = rp + (size_t)(b * cSO + t) * c3H;
    const float* ydr = yd + (size_t)(b * cSO + t) * c3H;
    float ir  = gc[j]          + pcr[j]          + rpr[j]          + ydr[j];
    float iz  = gc[cH + j]     + pcr[cH + j]     + rpr[cH + j]     + ydr[cH + j];
    float in_ = gc[2 * cH + j] + pcr[2 * cH + j] + rpr[2 * cH + j] + ydr[2 * cH + j];
    float r  = 1.0f / (1.0f + expf(-(ir + ar[v] + br)));
    float z  = 1.0f / (1.0f + expf(-(iz + az[v] + bz)));
    float nn = tanhf(in_ + r * (an_[v] + bn));
    float ho = (float)As[lr * cH + j];
    hn[(size_t)m * cH + j] = (h16)((1.0f - z) * nn + z * ho);
  }
}

// per-step loss: block per b, thread per y; logits + log_softmax + NLL accumulate
__global__ __launch_bounds__(512) void dec_loss_k(
    const h16* __restrict__ h, const float* __restrict__ oW, const float* __restrict__ ob,
    const float* __restrict__ y, int t, float* __restrict__ acc) {
  __shared__ float red[512];
  int b = blockIdx.x, yy = threadIdx.x;
  const h16* hp = h + (size_t)(b * cY + yy) * cH;
  float dot = ob[0];
  for (int j = 0; j < cH; ++j) dot += (float)hp[j] * oW[j];
  red[yy] = dot; __syncthreads();
  for (int o = 256; o; o >>= 1) { if (yy < o) red[yy] = fmaxf(red[yy], red[yy + o]); __syncthreads(); }
  float mx = red[0]; __syncthreads();
  red[yy] = expf(dot - mx); __syncthreads();
  for (int o = 256; o; o >>= 1) { if (yy < o) red[yy] += red[yy + o]; __syncthreads(); }
  float lse = mx + logf(red[0]); __syncthreads();
  red[yy] = -(dot - lse) * y[(size_t)(b * cSO + t) * cY + yy]; __syncthreads();
  for (int o = 256; o; o >>= 1) { if (yy < o) red[yy] += red[yy + o]; __syncthreads(); }
  if (yy == 0) atomicAdd(acc, red[0]);
}

__global__ void reduce_sum_k(const float* __restrict__ v, int n, float* __restrict__ acc) {
  __shared__ float red[256];
  float s = 0.0f;
  for (int i = blockIdx.x * 256 + threadIdx.x; i < n; i += gridDim.x * 256) s += v[i];
  red[threadIdx.x] = s; __syncthreads();
  for (int o = 128; o; o >>= 1) { if (threadIdx.x < o) red[threadIdx.x] += red[threadIdx.x + o]; __syncthreads(); }
  if (threadIdx.x == 0) atomicAdd(acc, red[0]);
}

__global__ void finalize_k(const float* __restrict__ acc, float* __restrict__ out) {
  if (threadIdx.x == 0) out[0] = acc[0] / acc[1];
}

// ---------------- host orchestration ----------------
static inline void gemm(hipStream_t s, const h16* A, int lda, long long sA,
                        const h16* W, int ldw, long long sW,
                        float* C, int ldc, long long sC, const float* bias,
                        int M, int N, int K, int batch, int accum, int act) {
  dim3 g(N / 128, M / 16, batch);   // all call sites: N%128==0, M%16==0, K%32==0, K<=512
  if (accum)
    gemm_nt_k<1, 0><<<g, 256, 0, s>>>(A, lda, sA, W, ldw, sW, C, ldc, sC, bias, M, N, K);
  else if (act)
    gemm_nt_k<0, 1><<<g, 256, 0, s>>>(A, lda, sA, W, ldw, sW, C, ldc, sC, bias, M, N, K);
  else
    gemm_nt_k<0, 0><<<g, 256, 0, s>>>(A, lda, sA, W, ldw, sW, C, ldc, sC, bias, M, N, K);
}
static inline void cvt(hipStream_t s, const float* src, h16* dst, int n) {
  f32tof16_k<<<(n + 255) / 256, 256, 0, s>>>(src, dst, n);
}

extern "C" void kernel_launch(void* const* d_in, const int* in_sizes, int n_in,
                              void* d_out, int out_size, void* d_ws, size_t ws_size,
                              hipStream_t stream) {
  (void)in_sizes; (void)n_in; (void)out_size; (void)ws_size;
  const int*   x      = (const int*)d_in[0];
  const int*   akey   = (const int*)d_in[1];
  const int*   skey   = (const int*)d_in[2];
  const int*   atype  = (const int*)d_in[3];
  const int*   croom  = (const int*)d_in[4];
  const float* yv     = (const float*)d_in[6];
  const int*   cfeat  = (const int*)d_in[8];
  const float* iemb   = (const float*)d_in[9];
  const float* attab  = (const float*)d_in[10];
  const float* cttab  = (const float*)d_in[11];
  const float* eWihF  = (const float*)d_in[12];
  const float* eWhhF  = (const float*)d_in[13];
  const float* ebihF  = (const float*)d_in[14];
  const float* ebhhF  = (const float*)d_in[15];
  const float* eWihB  = (const float*)d_in[16];
  const float* eWhhB  = (const float*)d_in[17];
  const float* ebihB  = (const float*)d_in[18];
  const float* ebhhB  = (const float*)d_in[19];
  const float* transW = (const float*)d_in[20];
  const float* transb = (const float*)d_in[21];
  const float* mergeW = (const float*)d_in[22];
  const float* mergeb = (const float*)d_in[23];
  const float* dWih   = (const float*)d_in[24];
  const float* dWhh   = (const float*)d_in[25];
  const float* dbih   = (const float*)d_in[26];
  const float* dbhh   = (const float*)d_in[27];
  const float* outW   = (const float*)d_in[28];
  const float* outb   = (const float*)d_in[29];
  float* out = (float*)d_out;

  char* base = (char*)d_ws;
  size_t off = 0;
  auto alloc = [&](size_t bytes) -> char* {
    char* p = base + off; off += (bytes + 255) & ~(size_t)255; return p;
  };
  float* EMB     = (float*)alloc((size_t)cB * cS * cE * 4);
  h16*   XS      = (h16*)  alloc((size_t)cS * cB * cE * 2);
  h16*   XSR     = (h16*)  alloc((size_t)cS * cB * cE * 2);
  h16*   COMB    = (h16*)  alloc((size_t)cB * cS * cE * 2);
  float* GI_F    = (float*)alloc((size_t)cS * cB * c3H * 4);
  float* GI_B    = (float*)alloc((size_t)cS * cB * c3H * 4);
  float* GICONST = GI_F;   // reuse after encoder scan (exactly 2x the GI buffer)
  float* OF_F    = (float*)alloc((size_t)cS * cB * cH * 4);
  float* OF_B    = (float*)alloc((size_t)cS * cB * cH * 4);
  h16*   EO16    = (h16*)  alloc((size_t)cB * cS * 2 * cH * 2);
  h16*   EOT16   = (h16*)  alloc((size_t)cB * 2 * cH * cS * 2);  // [B,2H,S]
  float* TRANS   = (float*)alloc((size_t)cB * cS * cE * 4);
  float* LOGITS  = (float*)alloc((size_t)cB * cY * cS * 4);
  h16*   ALPHA   = (h16*)  alloc((size_t)cB * cY * cS * 2);
  float* CTX     = (float*)alloc((size_t)cBY * 2 * cH * 4);
  h16*   ATT16   = (h16*)  alloc((size_t)cBY * 2 * cH * 2);
  h16*   SATT16  = (h16*)  alloc((size_t)cBY * 2 * cH * 2);
  h16*   ATB16   = (h16*)  alloc((size_t)cBY * 32 * 2);
  float* H0      = (float*)alloc((size_t)cB * cH * 4);
  float* PCNT    = (float*)alloc((size_t)4 * c3H * 4);
  h16*   ROOM16  = (h16*)  alloc((size_t)cB * cSO * cE * 2);
  float* RPROJ   = (float*)alloc((size_t)cB * cSO * c3H * 4);
  h16*   AT16    = (h16*)  alloc((size_t)cY * 32 * 2);
  h16*   AEMB16  = (h16*)  alloc((size_t)cY * cE * 2);
  h16*   SEMB16  = (h16*)  alloc((size_t)cY * cE * 2);
  float* KYF     = (float*)alloc((size_t)cY * c3H * 4);
  h16*   KYT16   = (h16*)  alloc((size_t)c3H * cY * 2);          // [768,512] transposed
  h16*   YP16    = (h16*)  alloc((size_t)cB * cSO * cY * 2);
  float* YDEP    = (float*)alloc((size_t)cB * cSO * c3H * 4);
  h16*   HA      = (h16*)  alloc((size_t)cBY * cH * 2);
  h16*   HB      = (h16*)  alloc((size_t)cBY * cH * 2);
  h16*   WihF16  = (h16*)  alloc((size_t)c3H * cE * 2);
  h16*   WhhF16  = (h16*)  alloc((size_t)c3H * cH * 2);
  h16*   WihB16  = (h16*)  alloc((size_t)c3H * cE * 2);
  h16*   WhhB16  = (h16*)  alloc((size_t)c3H * cH * 2);
  h16*   TW16    = (h16*)  alloc((size_t)cE * 2 * cH * 2);
  h16*   DWih16  = (h16*)  alloc((size_t)c3H * cDD * 2);
  h16*   DWhh16  = (h16*)  alloc((size_t)c3H * cH * 2);
  float* ACC     = (float*)alloc(256);

  zero2_k<<<1, 32, 0, stream>>>(ACC);

  // weight conversions to f16
  cvt(stream, eWihF, WihF16, c3H * cE);
  cvt(stream, eWhhF, WhhF16, c3H * cH);
  cvt(stream, eWihB, WihB16, c3H * cE);
  cvt(stream, eWhhB, WhhB16, c3H * cH);
  cvt(stream, transW, TW16, cE * 2 * cH);
  cvt(stream, dWih, DWih16, c3H * cDD);
  cvt(stream, dWhh, DWhh16, c3H * cH);

  // gathers
  gather16_k<<<(cY * 32 + 255) / 256, 256, 0, stream>>>(attab, atype, AT16, cY, 32);
  gather16_k<<<(cY * cE + 255) / 256, 256, 0, stream>>>(iemb, akey, AEMB16, cY, cE);
  gather16_k<<<(cY * cE + 255) / 256, 256, 0, stream>>>(iemb, skey, SEMB16, cY, cE);
  gather16_k<<<(cB * cSO * cE + 255) / 256, 256, 0, stream>>>(iemb, croom, ROOM16, cB * cSO, cE);

  // embeddings + time-major f16 copies
  build_emb_k<<<(cB * cS * cE + 255) / 256, 256, 0, stream>>>(iemb, x, EMB, XS, XSR);

  // GI = xs @ Wih^T + bih (both directions, fully parallel)
  gemm(stream, XS,  cE, 0, WihF16, cE, 0, GI_F, c3H, 0, ebihF, cS * cB, c3H, cE, 1, 0, 0);
  gemm(stream, XSR, cE, 0, WihB16, cE, 0, GI_B, c3H, 0, ebihB, cS * cB, c3H, cE, 1, 0, 0);

  // persistent bi-GRU scan (2 WGPs, LDS ping-pong, fused WMMA+gates)
  enc_scan_k<<<2, 256, 0, stream>>>(GI_F, GI_B, WhhF16, WhhB16, ebhhF, ebhhB, OF_F, OF_B);

  pack_enc_k<<<(cB * cS * 2 * cH + 255) / 256, 256, 0, stream>>>(OF_F, OF_B, EO16, EOT16);

  // trans_enc = tanh(enc_out @ trans_W^T + b); comb = emb + trans_enc (f16)
  gemm(stream, EO16, 2 * cH, 0, TW16, 2 * cH, 0, TRANS, cE, 0, transb, cB * cS, cE, 2 * cH, 1, 0, 1);
  comb_k<<<(cB * cS * cE + 255) / 256, 256, 0, stream>>>(EMB, TRANS, COMB);

  // attention for each key set: logits -> softmax -> context (all NT via EOT16)
  for (int key = 0; key < 2; ++key) {
    const h16* K16 = key ? SEMB16 : AEMB16;
    h16* DST = key ? SATT16 : ATT16;
    gemm(stream, K16, cE, 0, COMB, cE, (long long)cS * cE,
         LOGITS, cS, (long long)cY * cS, nullptr, cY, cS, cE, cB, 0, 0);
    softmax_rows_k<<<(cB * cY + 7) / 8, 256, 0, stream>>>(LOGITS, ALPHA, cB * cY, cS);
    gemm(stream, ALPHA, cS, (long long)cY * cS, EOT16, cS, (long long)2 * cH * cS,
         CTX, 2 * cH, (long long)cY * 2 * cH, nullptr, cY, 2 * cH, cS, cB, 0, 0);
    cvt(stream, CTX, DST, cBY * 2 * cH);
  }

  // merge -> h0 -> broadcast decoder state
  merge_k<<<cB, cH, 0, stream>>>(OF_F + (size_t)(cS - 1) * cB * cH,
                                 OF_B + (size_t)(cS - 1) * cB * cH, mergeW, mergeb, H0);
  hinit_k<<<(cBY * cH + 255) / 256, 256, 0, stream>>>(H0, HA);
  atb_k<<<(cBY * 32 + 255) / 256, 256, 0, stream>>>(AT16, ATB16);
  yp_k<<<(cB * cSO * cY + 255) / 256, 256, 0, stream>>>(yv, YP16);

  // gi_const = [att | at_b | satt] @ Wih-slices^T + bih   (column offsets 0/672/992)
  gemm(stream, ATT16,  2 * cH, 0, DWih16,       cDD, 0, GICONST, c3H, 0, dbih,    cBY, c3H, 2 * cH, 1, 0, 0);
  gemm(stream, ATB16,  32,     0, DWih16 + 672, cDD, 0, GICONST, c3H, 0, nullptr, cBY, c3H, 32,     1, 1, 0);
  gemm(stream, SATT16, 2 * cH, 0, DWih16 + 992, cDD, 0, GICONST, c3H, 0, nullptr, cBY, c3H, 2 * cH, 1, 1, 0);

  // counter / room / y-dependent projections (hoisted out of the scan)
  projcnt_k<<<(4 * c3H + 255) / 256, 256, 0, stream>>>(cttab, dWih, PCNT);
  gemm(stream, ROOM16, cE, 0, DWih16 + 544, cDD, 0, RPROJ, c3H, 0, nullptr, cB * cSO, c3H, cE, 1, 0, 0);
  gemm(stream, AT16,   32, 0, DWih16 + 704, cDD, 0, KYF, c3H, 0, nullptr, cY, c3H, 32, 1, 0, 0);
  gemm(stream, SEMB16, cE, 0, DWih16 + 736, cDD, 0, KYF, c3H, 0, nullptr, cY, c3H, cE, 1, 1, 0);
  gemm(stream, AEMB16, cE, 0, DWih16 + 864, cDD, 0, KYF, c3H, 0, nullptr, cY, c3H, cE, 1, 1, 0);
  transpose16_k<<<(cY * c3H + 255) / 256, 256, 0, stream>>>(KYF, KYT16, cY, c3H);
  gemm(stream, YP16, cY, 0, KYT16, cY, 0, YDEP, c3H, 0, nullptr, cB * cSO, c3H, cY, 1, 0, 0);

  // y.sum() for final normalization
  reduce_sum_k<<<256, 256, 0, stream>>>(yv, cB * cSO * cY, ACC + 1);

  // decoder scan: 32 steps of fused WMMA GRU + loss
  for (int t = 0; t < cSO; ++t) {
    const h16* hc = (t & 1) ? HB : HA;
    h16* hn = (t & 1) ? HA : HB;
    dec_step_k<<<(cBY / 16) * 2, 256, 0, stream>>>(hc, hn, DWhh16, dbhh, GICONST,
                                                   PCNT, RPROJ, YDEP, cfeat, t);
    dec_loss_k<<<cB, cY, 0, stream>>>(hn, outW, outb, yv, t, ACC);
  }

  finalize_k<<<1, 32, 0, stream>>>(ACC, out);
}
